// NeuralNetwork_28819230556388
// MI455X (gfx1250) — matrix operand order repaired and checked by reading, not compiled, not run on hardware
//
#include <hip/hip_runtime.h>

// ---------- types ----------
typedef __attribute__((ext_vector_type(16))) __bf16 v16bf;
typedef __attribute__((ext_vector_type(8)))  float  v8f;
typedef __attribute__((ext_vector_type(4)))  float  v4f;
typedef __attribute__((ext_vector_type(4)))  int    v4i;

struct __align__(16) U4 { unsigned int x, y, z, w; };
struct __align__(8)  U2 { unsigned int x, y; };
struct __align__(16) F4 { float x, y, z, w; };

// ---------- problem / tiling constants ----------
#define NB        262144          // batch rows
#define DD        64              // layer width
#define NMAT      15              // 8 next + 7 skip weight matrices
#define ROWSTRIDE 144             // bytes per 64-bf16 row in LDS (128 data + 16 pad)
#define MATBYTES  (64 * ROWSTRIDE)        // 9216 B per weight matrix in LDS
#define WLDS      (NMAT * MATBYTES)       // 138240 B of weights in LDS
#define RPW       32              // batch rows per wave (2 N-tiles)
#define NWAVE     4               // waves per workgroup
#define HBUF      (RPW * ROWSTRIDE)       // 4608 B per hidden buffer
#define LDSTOT    (WLDS + NWAVE * 3 * HBUF)  // 193536 B total

// ---------- async 16B global->LDS copy (ASYNCcnt-tracked, no VGPR data) ----------
__device__ __forceinline__ void async_cp16(const void* g, void* l) {
  __builtin_amdgcn_global_load_async_to_lds_b128(
      (__attribute__((address_space(1))) v4i*)(g),
      (__attribute__((address_space(3))) v4i*)(l),
      /*offset=*/0, /*cpol=*/0);
}

// ---------- WMMA wrapper ----------
__device__ __forceinline__ v8f wmma_bf16(v16bf a, v16bf b, v8f c) {
  // D = A(16x32 bf16) x B(32x16 bf16) + C(16x16 f32)
  return __builtin_amdgcn_wmma_f32_16x16x32_bf16(
      /*neg_a=*/false, a, /*neg_b=*/false, b,
      /*c_mod=*/(short)0, c, /*reuse_a=*/false, /*reuse_b=*/false);
}

// A fragment: weights, row M = mt*16 + l16; k-chunks at (L/16)*8 and 16+(L/16)*8
__device__ __forceinline__ v16bf load_afrag(const char* mat, int l16, int hl,
                                            int mt, int kc) {
  const char* p = mat + (mt * 16 + l16) * ROWSTRIDE + kc * 64 + hl * 16;
  union { v16bf v; U4 u[2]; } f;
  f.u[0] = *(const U4*)(p);
  f.u[1] = *(const U4*)(p + 32);
  return f.v;
}

// B fragment: activations, batch row N = nt*16 + l16; k-chunks at (L/16)*8 and 16+(L/16)*8
__device__ __forceinline__ v16bf load_bfrag(const char* hb, int l16, int hl,
                                            int nt, int kc) {
  const char* p = hb + (nt * 16 + l16) * ROWSTRIDE + kc * 64 + hl * 16;
  union { v16bf v; U4 u[2]; } f;
  f.u[0] = *(const U4*)(p);
  f.u[1] = *(const U4*)(p + 32);
  return f.v;
}

// one matmul term:  sum[nt][mt] (+)= relu( W @ h^T + bias )
__device__ __forceinline__ void mm_term(v8f sum[2][4], const char* wmat,
                                        const char* hsrc, const float* bias,
                                        bool first, int l16, int hl) {
  v16bf A[4][2];
#pragma unroll
  for (int mt = 0; mt < 4; ++mt)
#pragma unroll
    for (int kc = 0; kc < 2; ++kc)
      A[mt][kc] = load_afrag(wmat, l16, hl, mt, kc);

  v8f bfr[4];
#pragma unroll
  for (int mt = 0; mt < 4; ++mt) {
    const float* bp = bias + mt * 16 + hl * 8;   // feature m = mt*16 + hl*8 + r
    union { v8f v; F4 f[2]; } r;
    r.f[0] = *(const F4*)(bp);
    r.f[1] = *(const F4*)(bp + 4);
    bfr[mt] = r.v;
  }

#pragma unroll
  for (int nt = 0; nt < 2; ++nt) {
    v16bf B0 = load_bfrag(hsrc, l16, hl, nt, 0);
    v16bf B1 = load_bfrag(hsrc, l16, hl, nt, 1);
#pragma unroll
    for (int mt = 0; mt < 4; ++mt) {
      v8f acc = bfr[mt];
      acc = wmma_bf16(A[mt][0], B0, acc);
      acc = wmma_bf16(A[mt][1], B1, acc);
#pragma unroll
      for (int e = 0; e < 8; ++e) acc[e] = __builtin_fmaxf(acc[e], 0.0f);
      sum[nt][mt] = first ? acc : (sum[nt][mt] + acc);
    }
  }
}

// store f32 sum -> bf16 hidden buffer (row-major, feature-contiguous)
__device__ __forceinline__ void store_h(char* hdst, v8f sum[2][4], int l16, int hl) {
#pragma unroll
  for (int nt = 0; nt < 2; ++nt)
#pragma unroll
    for (int mt = 0; mt < 4; ++mt) {
      union { __bf16 h[8]; U4 u; } pk;
#pragma unroll
      for (int e = 0; e < 8; ++e) pk.h[e] = (__bf16)sum[nt][mt][e];
      // batch row = nt*16 + l16 ; feature base = mt*16 + hl*8  (bytes: mt*32 + hl*16)
      *(U4*)(hdst + (nt * 16 + l16) * ROWSTRIDE + mt * 32 + hl * 16) = pk.u;
    }
}

// ---------- kernel 1: f32 weights -> bf16 scratch (next[0..7] then skip[0..6]) ----
__global__ void cvt_weights_kernel(const float* __restrict__ wn,
                                   const float* __restrict__ wsk,
                                   unsigned short* __restrict__ out) {
  int i = blockIdx.x * 256 + threadIdx.x;
  if (i < NMAT * DD * DD) {
    float v = (i < 8 * DD * DD) ? wn[i] : wsk[i - 8 * DD * DD];
    __bf16 b = (__bf16)v;
    out[i] = __builtin_bit_cast(unsigned short, b);
  }
}

// ---------- kernel 2: fused layer chain ----------
__global__ __launch_bounds__(NWAVE * 32, 1) void mlp_chain_kernel(
    const float* __restrict__ x, const unsigned short* __restrict__ wbf,
    const float* __restrict__ bn, const float* __restrict__ bsk,
    float* __restrict__ out) {
  __shared__ __align__(16) char smem[LDSTOT];

  const int tid  = threadIdx.x;
  const int lane = tid & 31, wave = tid >> 5;
  const int l16  = lane & 15, hl = lane >> 4;
  const size_t Rbase = (size_t)blockIdx.x * (NWAVE * RPW) + (size_t)wave * RPW;

  char* hb = smem + WLDS + wave * (3 * HBUF);   // 3 rotating hidden buffers / wave

  // --- async stage: all 15 bf16 weight matrices -> padded LDS rows ---
  // GLOBAL_LOAD_ASYNC_TO_LDS_B128: memory -> LDS direct, tracked by ASYNCcnt,
  // no VGPR data movement. 16B per lane per op.
  {
    const char* src = (const char*)wbf;
    for (int i = tid; i < NMAT * 64 * 8; i += NWAVE * 32) {
      int row = i >> 3, q = i & 7;              // 8 x 16B chunks per 128B row
      async_cp16(src + row * 128 + q * 16,
                 smem + row * ROWSTRIDE + q * 16);
    }
  }

  // --- stage x tile -> h0 (slot 0), f32 -> bf16, row-major (non-temporal reads) ---
  {
    char* h0 = hb;
#pragma unroll
    for (int it = 0; it < 16; ++it) {
      int r = it * 2 + hl;                      // 0..31
      int c = l16 * 4;                          // 0..60
      v4f xv = __builtin_nontemporal_load(
          (const v4f*)(x + (Rbase + (size_t)r) * DD + c));
      union { __bf16 h[4]; U2 u; } pk;
      pk.h[0] = (__bf16)xv.x; pk.h[1] = (__bf16)xv.y;
      pk.h[2] = (__bf16)xv.z; pk.h[3] = (__bf16)xv.w;
      *(U2*)(h0 + r * ROWSTRIDE + c * 2) = pk.u;
    }
  }

  __builtin_amdgcn_s_wait_asynccnt(0);          // weights landed in LDS
  __syncthreads();

  v8f sum[2][4];

  // j = 1 : h1 = relu(h0 @ Wn0^T + bn0)          (slot 1)
  mm_term(sum, smem + 0 * MATBYTES, hb + 0 * HBUF, bn + 0 * DD, true, l16, hl);
  store_h(hb + 1 * HBUF, sum, l16, hl);

  // j = 2..6 : h_j = relu(h_{j-1} Wn_{j-1}) + relu(h_{j-2} Ws_{j-2})
#pragma unroll
  for (int j = 2; j <= 6; ++j) {
    mm_term(sum, smem + (j - 1) * MATBYTES, hb + ((j - 1) % 3) * HBUF,
            bn + (j - 1) * DD, true, l16, hl);
    mm_term(sum, smem + (8 + j - 2) * MATBYTES, hb + ((j - 2) % 3) * HBUF,
            bsk + (j - 2) * DD, false, l16, hl);
    store_h(hb + (j % 3) * HBUF, sum, l16, hl);
  }

  // j = 7 : final output (returned buffer) -> global f32 (non-temporal)
  mm_term(sum, smem + 6 * MATBYTES, hb + (6 % 3) * HBUF, bn + 6 * DD, true, l16, hl);
  mm_term(sum, smem + (8 + 5) * MATBYTES, hb + (5 % 3) * HBUF, bsk + 5 * DD, false, l16, hl);
#pragma unroll
  for (int nt = 0; nt < 2; ++nt)
#pragma unroll
    for (int mt = 0; mt < 4; ++mt) {
      float* p = out + (Rbase + (size_t)(nt * 16 + l16)) * DD + mt * 16 + hl * 8;
      union { v8f v; v4f f[2]; } r; r.v = sum[nt][mt];
      __builtin_nontemporal_store(r.f[0], (v4f*)(p));
      __builtin_nontemporal_store(r.f[1], (v4f*)(p + 4));
    }
}

// ---------- host entry ----------
extern "C" void kernel_launch(void* const* d_in, const int* in_sizes, int n_in,
                              void* d_out, int out_size, void* d_ws, size_t ws_size,
                              hipStream_t stream) {
  const float* x   = (const float*)d_in[0];
  const float* wn  = (const float*)d_in[1];   // Ws_next  (8,64,64)
  const float* bn  = (const float*)d_in[2];   // bs_next  (8,64)
  const float* wsk = (const float*)d_in[3];   // Ws_skip  (7,64,64)
  const float* bsk = (const float*)d_in[4];   // bs_skip  (7,64)
  float* out = (float*)d_out;
  unsigned short* wbf = (unsigned short*)d_ws;  // bf16 weights, 122880 B

  cvt_weights_kernel<<<(NMAT * DD * DD + 255) / 256, 256, 0, stream>>>(wn, wsk, wbf);
  mlp_chain_kernel<<<NB / (NWAVE * RPW), NWAVE * 32, 0, stream>>>(x, wbf, bn, bsk, out);
}
